// MultiheadAttentionCore_22754736734239
// MI455X (gfx1250) — matrix operand order repaired and checked by reading, not compiled, not run on hardware
//
#include <hip/hip_runtime.h>
#include <hip/hip_bf16.h>

typedef __bf16 bf16_t;
typedef bf16_t v8bf  __attribute__((ext_vector_type(8)));
typedef bf16_t v16bf __attribute__((ext_vector_type(16)));
typedef float  v8f   __attribute__((ext_vector_type(8)));

#define B_  2
#define S_  2048
#define D_  1024
#define H_  16
#define HD_ 64
#define MTOT (B_ * S_)   // 4096

__device__ __forceinline__ v16bf ldfrag_c(const bf16_t* p) {
  // 16 contiguous bf16 (two 16B loads)
  v8bf lo = *(const v8bf*)(p);
  v8bf hi = *(const v8bf*)(p + 8);
  return __builtin_shufflevector(lo, hi, 0,1,2,3,4,5,6,7,8,9,10,11,12,13,14,15);
}
__device__ __forceinline__ v16bf ldfrag_g16(const bf16_t* p) {
  // two 8-element chunks 16 elements apart (A-fragment pattern)
  v8bf lo = *(const v8bf*)(p);
  v8bf hi = *(const v8bf*)(p + 16);
  return __builtin_shufflevector(lo, hi, 0,1,2,3,4,5,6,7,8,9,10,11,12,13,14,15);
}
__device__ __forceinline__ v8f wmma_bf16(v16bf a, v16bf b, v8f c) {
  return __builtin_amdgcn_wmma_f32_16x16x32_bf16(false, a, false, b, (short)0, c, false, false);
}
__device__ __forceinline__ float half16_max(float v) {
  #pragma unroll
  for (int m = 1; m < 16; m <<= 1) v = fmaxf(v, __shfl_xor(v, m, 32));
  return v;
}
__device__ __forceinline__ float half16_sum(float v) {
  #pragma unroll
  for (int m = 1; m < 16; m <<= 1) v += __shfl_xor(v, m, 32);
  return v;
}

// ---------------------------------------------------------------- converts
__global__ void cvt_f32_bf16(const float* __restrict__ src, bf16_t* __restrict__ dst, int n) {
  int i = blockIdx.x * 256 + threadIdx.x;
  if (i < n) dst[i] = (bf16_t)src[i];
}

// ---------------------------------------------------------------- RoPE
// src/dst layout [B,H,S,HD] bf16
__global__ void rope_kernel(const bf16_t* __restrict__ src, bf16_t* __restrict__ dst, int n) {
  int i = blockIdx.x * 256 + threadIdx.x;
  if (i >= n) return;
  int hd = i & (HD_ - 1);
  int s  = (i >> 6) & (S_ - 1);
  int d2 = hd & 31;
  // inv_freq = ROPE_BASE^(-2*d2/HD)
  float inv_freq = __expf(-(float)(2 * d2) * (1.0f / 64.0f) * 9.210340371976184f); // ln(10000)
  float ang = (float)s * inv_freq;
  float sn, cs;
  __sincosf(ang, &sn, &cs);
  float xv = (float)src[i];
  int pair = (hd < 32) ? (i + 32) : (i - 32);
  float pv = (float)src[pair];
  float rot = (hd < 32) ? -pv : pv;
  dst[i] = (bf16_t)(xv * cs + rot * sn);
}

// ---------------------------------------------------------------- WMMA GEMM
// OUT = A[M=4096,K=1024] @ W[N=1024,K=1024]^T + bias
// Block = 4 waves: 64 M-rows x 64 N-cols. A/W tiles staged to LDS with
// async copies (double buffered, ASYNCcnt), fragments read via ds_load_b128.
// mode 0: bf16 out, head layout      [B,H,S,HD]
// mode 1: bf16 out, head transposed  [B,H,HD,S]   (for V)
// mode 2: f32 out, flat [M,N]                      (final projection)
#define STG_ELEMS 4096            // per-stage elements: A(64x32)=2048 + W(64x32)=2048
__global__ __launch_bounds__(128) void gemm_wmma(
    const bf16_t* __restrict__ A, const bf16_t* __restrict__ W,
    const float* __restrict__ bias, void* __restrict__ outp, int mode) {
  const int K = D_;
  __shared__ bf16_t smem[2 * STG_ELEMS];   // 16 KB, double buffered

  // Make the LDS array escape so alias analysis treats the inline-asm async
  // copies as potential writers of smem (otherwise the ds_load fragments get
  // folded to undef since no compiler-visible store exists).
  asm volatile("" :: "v"(smem) : "memory");

  int tid  = threadIdx.x;
  int wid  = tid >> 5;
  int lane = tid & 31;
  int l15  = lane & 15;
  int hi   = lane >> 4;
  int mBlk  = blockIdx.x >> 4;   // 64 M-blocks of 64 rows
  int nQuad = blockIdx.x & 15;   // 16 N-quads of 64 cols
  int M0b = mBlk << 6;
  int N0  = nQuad << 6;
  int kbA = hi ? 16 : 0;
  int kbB = hi ? 16 : 0;

  // per-thread copy chunks: 256 16B-chunks per tile, 2 per thread
  int c0  = tid * 2;

  auto stage_copy = [&](int s, int k0) {
    unsigned baseA = (unsigned)(s * STG_ELEMS) * 2u;       // LDS byte offsets (smem @ 0)
    unsigned baseW = baseA + 2048u * 2u;
    const bf16_t* gA = A + (size_t)M0b * K + k0;
    const bf16_t* gW = W + (size_t)N0  * K + k0;
    #pragma unroll
    for (int i = 0; i < 2; ++i) {
      int c   = c0 + i;
      int row = c >> 2;            // 4 chunks per 32-elem row
      int col = (c & 3) * 8;       // element offset
      const bf16_t* ga = gA + (size_t)row * K + col;
      unsigned la = baseA + (unsigned)(row * 32 + col) * 2u;
      asm volatile("global_load_async_to_lds_b128 %0, %1, off"
                   :: "v"(la), "v"(ga), "v"(smem) : "memory");
      const bf16_t* gw = gW + (size_t)row * K + col;
      unsigned lw = baseW + (unsigned)(row * 32 + col) * 2u;
      asm volatile("global_load_async_to_lds_b128 %0, %1, off"
                   :: "v"(lw), "v"(gw), "v"(smem) : "memory");
    }
  };

  v8f acc[4];
  #pragma unroll
  for (int t = 0; t < 4; ++t) acc[t] = (v8f)0.0f;

  stage_copy(0, 0);
  for (int k0 = 0; k0 < K; k0 += 32) {
    int cur = (k0 >> 5) & 1;
    if (k0 + 32 < K) {
      stage_copy(cur ^ 1, k0 + 32);
      asm volatile("s_wait_asynccnt 0x4" ::: "memory");   // stage `cur` landed
    } else {
      asm volatile("s_wait_asynccnt 0x0" ::: "memory");
    }
    __syncthreads();
    const bf16_t* sa = smem + cur * STG_ELEMS + ((wid * 16 + l15) * 32 + kbA);
    v16bf a = ldfrag_c(sa);
    const bf16_t* sw = smem + cur * STG_ELEMS + 2048;
    #pragma unroll
    for (int t = 0; t < 4; ++t) {
      v16bf b = ldfrag_c(sw + (t * 16 + l15) * 32 + kbB);
      acc[t] = wmma_bf16(a, b, acc[t]);
    }
    __syncthreads();
  }

  int M0 = M0b + wid * 16;
  #pragma unroll
  for (int t = 0; t < 4; ++t) {
    #pragma unroll
    for (int r = 0; r < 8; ++r) {
      int m = M0 + r + hi * 8;
      int n = N0 + t * 16 + l15;
      float v = acc[t][r] + bias[n];
      if (mode == 0) {
        int b = m >> 11, s = m & (S_ - 1), h = n >> 6, hd = n & 63;
        ((bf16_t*)outp)[(((size_t)(b * H_ + h) * S_) + s) * HD_ + hd] = (bf16_t)v;
      } else if (mode == 1) {
        int b = m >> 11, s = m & (S_ - 1), h = n >> 6, hd = n & 63;
        ((bf16_t*)outp)[(((size_t)(b * H_ + h) * HD_) + hd) * S_ + s] = (bf16_t)v;
      } else {
        ((float*)outp)[(size_t)m * D_ + n] = v;
      }
    }
  }
}

// ---------------------------------------------------------------- flash attention
// Q,K: [B,H,S,64] bf16 (roped); VT: [B,H,64,S] bf16; CTX out: [B,S,D] bf16
__global__ __launch_bounds__(128) void attn_wmma(
    const bf16_t* __restrict__ Q, const bf16_t* __restrict__ Km,
    const bf16_t* __restrict__ VT, bf16_t* __restrict__ CTX) {
  __shared__ bf16_t ldsP[4][16 * 32];
  int widInB = threadIdx.x >> 5;
  int wave = blockIdx.x * 4 + widInB;
  int lane = threadIdx.x & 31, l15 = lane & 15, hi = lane >> 4;
  int b   = wave >> 11;            // / (H * S/16)
  int rem = wave & 2047;
  int h   = rem >> 7;
  int qT  = rem & 127;
  int q0  = qT << 4;
  size_t bh = (size_t)(b * H_ + h);
  const bf16_t* Qh = Q  + bh * S_ * HD_;
  const bf16_t* Kh = Km + bh * S_ * HD_;
  const bf16_t* Vh = VT + bh * HD_ * S_;

  int kbA = hi ? 16 : 0;
  int kbB = hi ? 16 : 0;
  int qrow = q0 + l15;
  v16bf aq0 = ldfrag_c(Qh + (size_t)qrow * HD_ + kbA);
  v16bf aq1 = ldfrag_c(Qh + (size_t)qrow * HD_ + 32 + kbA);

  v8f o[4];
  #pragma unroll
  for (int t = 0; t < 4; ++t) o[t] = (v8f)0.0f;
  float rmax[8], rsum[8];
  #pragma unroll
  for (int r = 0; r < 8; ++r) { rmax[r] = -__builtin_inff(); rsum[r] = 0.0f; }
  const float sc = 0.125f;   // 1/sqrt(64)

  for (int kv0 = 0; kv0 < q0 + 16; kv0 += 32) {
    // QK^T: two 16-key tiles, K-dim 64 = 2 wmma steps each
    v16bf b00 = ldfrag_c(Kh + (size_t)(kv0 + l15) * HD_ + kbB);
    v16bf b01 = ldfrag_c(Kh + (size_t)(kv0 + l15) * HD_ + 32 + kbB);
    v16bf b10 = ldfrag_c(Kh + (size_t)(kv0 + 16 + l15) * HD_ + kbB);
    v16bf b11 = ldfrag_c(Kh + (size_t)(kv0 + 16 + l15) * HD_ + 32 + kbB);
    v8f s0 = (v8f)0.0f, s1 = (v8f)0.0f;
    s0 = wmma_bf16(aq0, b00, s0);
    s0 = wmma_bf16(aq1, b01, s0);
    s1 = wmma_bf16(aq0, b10, s1);
    s1 = wmma_bf16(aq1, b11, s1);

    // online softmax over this 32-key block
    #pragma unroll
    for (int r = 0; r < 8; ++r) {
      int row  = q0 + r + hi * 8;
      int key0 = kv0 + l15;
      float e0 = (key0      <= row) ? s0[r] * sc : -__builtin_inff();
      float e1 = (key0 + 16 <= row) ? s1[r] * sc : -__builtin_inff();
      float mx   = half16_max(fmaxf(e0, e1));
      float mnew = fmaxf(rmax[r], mx);
      float corr = __expf(rmax[r] - mnew);
      float p0 = __expf(e0 - mnew);
      float p1 = __expf(e1 - mnew);
      float rs = half16_sum(p0 + p1);
      rsum[r] = rsum[r] * corr + rs;
      rmax[r] = mnew;
      #pragma unroll
      for (int t = 0; t < 4; ++t) o[t][r] *= corr;
      int rr = r + hi * 8;
      ldsP[widInB][rr * 32 + l15]      = (bf16_t)p0;
      ldsP[widInB][rr * 32 + l15 + 16] = (bf16_t)p1;
    }
    asm volatile("s_wait_dscnt 0" ::: "memory");

    // P (16x32) as A-fragment from LDS, then P @ V
    v16bf pa = ldfrag_c(&ldsP[widInB][l15 * 32 + kbA]);
    #pragma unroll
    for (int t = 0; t < 4; ++t) {
      const bf16_t* vp = Vh + (size_t)(t * 16 + l15) * S_ + kv0 + kbB;
      v16bf vb = ldfrag_c(vp);
      o[t] = wmma_bf16(pa, vb, o[t]);
    }
  }

  // normalize + write ctx in [B,S,D] layout
  #pragma unroll
  for (int r = 0; r < 8; ++r) {
    float inv = 1.0f / rsum[r];
    int row = q0 + r + hi * 8;
    #pragma unroll
    for (int t = 0; t < 4; ++t) {
      int col = h * 64 + t * 16 + l15;
      CTX[(size_t)(b * S_ + row) * D_ + col] = (bf16_t)(o[t][r] * inv);
    }
  }
}

// ---------------------------------------------------------------- launch
extern "C" void kernel_launch(void* const* d_in, const int* in_sizes, int n_in,
                              void* d_out, int out_size, void* d_ws, size_t ws_size,
                              hipStream_t stream) {
  const float* x  = (const float*)d_in[0];
  // d_in[1] = attn_mask (causal tril) -- implied by kernel, unused
  const float* Wq = (const float*)d_in[2];
  const float* bq = (const float*)d_in[3];
  const float* Wk = (const float*)d_in[4];
  const float* bk = (const float*)d_in[5];
  const float* Wv = (const float*)d_in[6];
  const float* bv = (const float*)d_in[7];
  const float* Wo = (const float*)d_in[8];
  const float* bo = (const float*)d_in[9];
  float* out = (float*)d_out;

  char* ws = (char*)d_ws;
  size_t off = 0;
  auto carve = [&](size_t bytes) -> void* {
    void* p = ws + off;
    off += (bytes + 255) & ~(size_t)255;
    return p;
  };
  const size_t nBig = (size_t)MTOT * D_;      // 4M elements
  const size_t nW   = (size_t)D_ * D_;        // 1M elements
  bf16_t* XB   = (bf16_t*)carve(nBig * 2);
  bf16_t* WQB  = (bf16_t*)carve(nW * 2);
  bf16_t* WKB  = (bf16_t*)carve(nW * 2);
  bf16_t* WVB  = (bf16_t*)carve(nW * 2);
  bf16_t* WOB  = (bf16_t*)carve(nW * 2);
  bf16_t* QRAW = (bf16_t*)carve(nBig * 2);
  bf16_t* KRAW = (bf16_t*)carve(nBig * 2);
  bf16_t* Qr   = (bf16_t*)carve(nBig * 2);
  bf16_t* Kr   = (bf16_t*)carve(nBig * 2);
  bf16_t* VTr  = (bf16_t*)carve(nBig * 2);
  bf16_t* CTX  = (bf16_t*)carve(nBig * 2);

  // 1) downcast inputs
  cvt_f32_bf16<<<(int)((nBig + 255) / 256), 256, 0, stream>>>(x, XB, (int)nBig);
  cvt_f32_bf16<<<(int)((nW + 255) / 256), 256, 0, stream>>>(Wq, WQB, (int)nW);
  cvt_f32_bf16<<<(int)((nW + 255) / 256), 256, 0, stream>>>(Wk, WKB, (int)nW);
  cvt_f32_bf16<<<(int)((nW + 255) / 256), 256, 0, stream>>>(Wv, WVB, (int)nW);
  cvt_f32_bf16<<<(int)((nW + 255) / 256), 256, 0, stream>>>(Wo, WOB, (int)nW);

  // 2) QKV projections (1024 blocks x 4 waves, 64x64 tiles, async-LDS staged)
  gemm_wmma<<<1024, 128, 0, stream>>>(XB, WQB, bq, QRAW, 0);
  gemm_wmma<<<1024, 128, 0, stream>>>(XB, WKB, bk, KRAW, 0);
  gemm_wmma<<<1024, 128, 0, stream>>>(XB, WVB, bv, VTr, 1);

  // 3) RoPE on Q, K
  rope_kernel<<<(int)((nBig + 255) / 256), 256, 0, stream>>>(QRAW, Qr, (int)nBig);
  rope_kernel<<<(int)((nBig + 255) / 256), 256, 0, stream>>>(KRAW, Kr, (int)nBig);

  // 4) causal flash attention
  attn_wmma<<<1024, 128, 0, stream>>>(Qr, Kr, VTr, CTX);

  // 5) output projection -> f32 d_out
  gemm_wmma<<<1024, 128, 0, stream>>>(CTX, WOB, bo, out, 2);
}
